// SoftMaxClassifier_18090402250920
// MI455X (gfx1250) — matrix operand order repaired and checked by reading, not compiled, not run on hardware
//
#include <hip/hip_runtime.h>
#include <hip/hip_bf16.h>
#include <math.h>

// ---------------- problem constants ----------------
#define R_PROP   20000
#define D_FEAT   1024
#define K_CLS    60          // foreground classes
#define KP1      61          // with background
#define TOPK     100
#define NEG_SENT (-1e30f)
#define IMG_W    1333.0f
#define IMG_H    800.0f

// workspace layout (bytes)
#define WS_SCORES_OFF 0                         // R*K floats = 4,800,000 B
#define WS_BX_OFF     4800000                   // R*4 floats = 320,000 B
#define WS_BFRAG_OFF  5120000                   // 4*32*32*16 halves = 131,072 B

typedef __attribute__((ext_vector_type(16))) _Float16 v16h;
typedef __attribute__((ext_vector_type(8)))  float    v8f;

// =====================================================================
// Prep: pack W into per-lane WMMA B-fragments (f16) + clip boxes.
// B layout (16-bit B 32x16, wave32): lane -> N = lane%16; half = lane/16;
// element e: p=e>>1, lo=e&1 -> K = ks*32 + half*8 + ((2p+lo)&7) + 16*((2p+lo)>>3).
// Flat index = (((nt*32 + ks)*32 + lane)*16 + e)  == idx below.
// =====================================================================
__global__ __launch_bounds__(256) void prep_kernel(
    const float* __restrict__ W, const float* __restrict__ boxes,
    _Float16* __restrict__ Bfrag, float* __restrict__ bx)
{
    int idx = blockIdx.x * 256 + threadIdx.x;
    if (idx < 4 * 32 * 32 * 16) {
        int e    = idx & 15;
        int lane = (idx >> 4) & 31;
        int ks   = (idx >> 9) & 31;
        int nt   = idx >> 14;
        int half = lane >> 4;
        int p    = e >> 1;
        int lo   = e & 1;
        int K    = ks * 32 + half * 8 + ((2 * p + lo) & 7) + (((2 * p + lo) >> 3) << 4);
        int N    = nt * 16 + (lane & 15);
        float v  = (N < KP1) ? W[(size_t)K * KP1 + N] : 0.0f;
        Bfrag[idx] = (_Float16)v;
    } else {
        int r = idx - 4 * 32 * 32 * 16;
        if (r < R_PROP) {
            float x1 = fminf(fmaxf(boxes[r * 4 + 0], 0.0f), IMG_W);
            float y1 = fminf(fmaxf(boxes[r * 4 + 1], 0.0f), IMG_H);
            float x2 = fminf(fmaxf(boxes[r * 4 + 2], 0.0f), IMG_W);
            float y2 = fminf(fmaxf(boxes[r * 4 + 3], 0.0f), IMG_H);
            bx[r * 4 + 0] = x1; bx[r * 4 + 1] = y1;
            bx[r * 4 + 2] = x2; bx[r * 4 + 3] = y2;
        }
    }
}

// =====================================================================
// GEMM (WMMA f16 -> f32) + softmax + threshold.
// Block = 128 threads = 4 waves; wave w computes the 16x16 tile
// [rows m0..m0+15] x [cols 16w..16w+15]; K swept in 32 steps of 32.
// =====================================================================
__global__ __launch_bounds__(128) void gemm_softmax_kernel(
    const float* __restrict__ feats, const float* __restrict__ bias,
    const _Float16* __restrict__ Bfrag, float* __restrict__ scores)
{
    __shared__ float lds_logits[16 * 64];

    const int tid  = threadIdx.x;
    const int wave = tid >> 5;          // == ntile (0..3)
    const int lane = tid & 31;
    const int half = lane >> 4;
    const int lcol = lane & 15;
    const int m0   = blockIdx.x * 16;
    const int row  = m0 + lcol;

    const float*    arow = feats + (size_t)row * D_FEAT;
    const _Float16* bptr = Bfrag + ((size_t)wave * 32 * 32 + lane) * 16;

    v8f acc = {};
    #pragma unroll 4
    for (int ks = 0; ks < 32; ++ks) {
        const int kb = ks * 32 + half * 8;
        if (ks + 1 < 32)
            __builtin_prefetch(arow + kb + 32, 0, 3);   // global_prefetch_b8

        // A fragment: 16-bit A 16x32 layout = two contiguous 8-float runs
        float4 f0 = *(const float4*)(arow + kb);
        float4 f1 = *(const float4*)(arow + kb + 4);
        float4 f2 = *(const float4*)(arow + kb + 16);
        float4 f3 = *(const float4*)(arow + kb + 20);
        v16h a;
        a[0]  = (_Float16)f0.x; a[1]  = (_Float16)f0.y;
        a[2]  = (_Float16)f0.z; a[3]  = (_Float16)f0.w;
        a[4]  = (_Float16)f1.x; a[5]  = (_Float16)f1.y;
        a[6]  = (_Float16)f1.z; a[7]  = (_Float16)f1.w;
        a[8]  = (_Float16)f2.x; a[9]  = (_Float16)f2.y;
        a[10] = (_Float16)f2.z; a[11] = (_Float16)f2.w;
        a[12] = (_Float16)f3.x; a[13] = (_Float16)f3.y;
        a[14] = (_Float16)f3.z; a[15] = (_Float16)f3.w;

        v16h bb = *(const v16h*)(bptr + (size_t)ks * 32 * 16);

        acc = __builtin_amdgcn_wmma_f32_16x16x32_f16(
            /*neg_a=*/false, a, /*neg_b=*/false, bb,
            /*c_mod=*/(short)0, acc, /*reuse_a=*/false, /*reuse_b=*/false);
    }

    // C layout: acc[v] holds (M = v + 8*half, N = lcol) of the 16x16 tile
    #pragma unroll
    for (int v = 0; v < 8; ++v) {
        int r = v + 8 * half;
        int c = wave * 16 + lcol;
        float lg = acc[v] + ((c < KP1) ? bias[c] : 0.0f);
        lds_logits[r * 64 + c] = lg;
    }
    __syncthreads();

    // row softmax over 61 classes, drop background, threshold -> scores
    if (tid < 16) {
        const float* lrow = lds_logits + tid * 64;
        float mx = lrow[0];
        #pragma unroll 1
        for (int c = 1; c < KP1; ++c) mx = fmaxf(mx, lrow[c]);
        float sum = 0.0f;
        #pragma unroll 1
        for (int c = 0; c < KP1; ++c) sum += __expf(lrow[c] - mx);
        float inv = 1.0f / sum;
        size_t base = (size_t)(m0 + tid) * K_CLS;
        #pragma unroll 1
        for (int c = 0; c < K_CLS; ++c) {
            float p = __expf(lrow[c] - mx) * inv;
            scores[base + c] = (p > 0.02f) ? p : NEG_SENT;
        }
    }
}

// =====================================================================
// Greedy per-class NMS, 100 sequential picks. Single 1024-thread WG.
// Scores array (R*K, L2-resident) scanned each round; IoU is class-
// agnostic so suppression only touches column c of iou>0.5 rows.
// =====================================================================
__global__ __launch_bounds__(1024) void nms_kernel(
    float* __restrict__ scores, const float* __restrict__ bx,
    float* __restrict__ out)
{
    __shared__ float s_val[32];
    __shared__ int   s_idx[32];
    __shared__ float s_box[4];
    __shared__ int   s_cls;
    __shared__ int   s_valid;

    const int tid  = threadIdx.x;
    const int lane = tid & 31;
    const int wv   = tid >> 5;
    const int N    = R_PROP * K_CLS;

    for (int it = 0; it < TOPK; ++it) {
        // ---- global argmax (first-index tie-break, matching jnp.argmax) ----
        float bv = -3.0e38f;
        int   bi = 0x7fffffff;
        for (int i = tid; i < N; i += 1024) {
            float v = scores[i];
            if (v > bv || (v == bv && i < bi)) { bv = v; bi = i; }
        }
        #pragma unroll
        for (int off = 16; off > 0; off >>= 1) {
            float ov = __shfl_xor(bv, off, 32);
            int   oi = __shfl_xor(bi, off, 32);
            if (ov > bv || (ov == bv && oi < bi)) { bv = ov; bi = oi; }
        }
        if (lane == 0) { s_val[wv] = bv; s_idx[wv] = bi; }
        __syncthreads();

        if (wv == 0) {
            float bv2 = s_val[lane];
            int   bi2 = s_idx[lane];
            #pragma unroll
            for (int off = 16; off > 0; off >>= 1) {
                float ov = __shfl_xor(bv2, off, 32);
                int   oi = __shfl_xor(bi2, off, 32);
                if (ov > bv2 || (ov == bv2 && oi < bi2)) { bv2 = ov; bi2 = oi; }
            }
            if (lane == 0) {
                int valid = (bv2 > NEG_SENT * 0.5f) ? 1 : 0;
                s_valid = valid;
                int r = bi2 / K_CLS;
                int c = bi2 - r * K_CLS;
                s_cls = c;
                float x1 = bx[r * 4 + 0], y1 = bx[r * 4 + 1];
                float x2 = bx[r * 4 + 2], y2 = bx[r * 4 + 3];
                s_box[0] = x1; s_box[1] = y1; s_box[2] = x2; s_box[3] = y2;
                if (valid) scores[bi2] = NEG_SENT;
                // outputs: boxes [0,400) | scores [400,500) | classes [500,600)
                out[it * 4 + 0] = valid ? x1 : 0.0f;
                out[it * 4 + 1] = valid ? y1 : 0.0f;
                out[it * 4 + 2] = valid ? x2 : 0.0f;
                out[it * 4 + 3] = valid ? y2 : 0.0f;
                out[400 + it]   = valid ? bv2 : 0.0f;
                out[500 + it]   = valid ? (float)c : -1.0f;
            }
        }
        __syncthreads();

        // ---- suppression: per-row IoU vs picked box, picked class column ----
        if (s_valid) {
            const float px1 = s_box[0], py1 = s_box[1];
            const float px2 = s_box[2], py2 = s_box[3];
            const float pa  = (px2 - px1) * (py2 - py1);
            const int   c   = s_cls;
            for (int r = tid; r < R_PROP; r += 1024) {
                float4 b = *(const float4*)(bx + r * 4);
                float ix1 = fmaxf(px1, b.x);
                float iy1 = fmaxf(py1, b.y);
                float ix2 = fminf(px2, b.z);
                float iy2 = fminf(py2, b.w);
                float inter = fmaxf(ix2 - ix1, 0.0f) * fmaxf(iy2 - iy1, 0.0f);
                float a2 = (b.z - b.x) * (b.w - b.y);
                float iou = inter / fmaxf(pa + a2 - inter, 1e-9f);
                if (iou > 0.5f) scores[r * K_CLS + c] = NEG_SENT;
            }
        }
        __threadfence_block();
        __syncthreads();
    }
}

// =====================================================================
extern "C" void kernel_launch(void* const* d_in, const int* in_sizes, int n_in,
                              void* d_out, int out_size, void* d_ws, size_t ws_size,
                              hipStream_t stream) {
    const float* feats = (const float*)d_in[0];   // [R, D]
    const float* W     = (const float*)d_in[1];   // [D, K+1]
    const float* bias  = (const float*)d_in[2];   // [K+1]
    const float* boxes = (const float*)d_in[3];   // [R, 4]
    float* out = (float*)d_out;

    char* ws = (char*)d_ws;
    float*    scores = (float*)(ws + WS_SCORES_OFF);
    float*    bx     = (float*)(ws + WS_BX_OFF);
    _Float16* Bfrag  = (_Float16*)(ws + WS_BFRAG_OFF);

    (void)in_sizes; (void)n_in; (void)out_size; (void)ws_size;

    // 1) pack W into WMMA B-fragments + clip boxes
    int prep_items = 4 * 32 * 32 * 16 + R_PROP;
    prep_kernel<<<(prep_items + 255) / 256, 256, 0, stream>>>(W, boxes, Bfrag, bx);

    // 2) WMMA GEMM + softmax + threshold  (1250 row-tiles of 16)
    gemm_softmax_kernel<<<R_PROP / 16, 128, 0, stream>>>(feats, bias, Bfrag, scores);

    // 3) sequential greedy NMS + topk (single persistent workgroup)
    nms_kernel<<<1, 1024, 0, stream>>>(scores, bx, out);
}
